// UniformMoERouter3_38165079392679
// MI455X (gfx1250) — compile-verified, matrix operand order checked by reading, not run on hardware
//
#include <hip/hip_runtime.h>
#include <hip/hip_bf16.h>

typedef __attribute__((ext_vector_type(16))) __bf16        v16bf;
typedef __attribute__((ext_vector_type(8)))  __bf16        v8bf;
typedef __attribute__((ext_vector_type(8)))  float         v8f;
typedef __attribute__((ext_vector_type(4)))  unsigned int  v4u;
typedef __attribute__((ext_vector_type(8)))  int           v8i;
typedef __attribute__((ext_vector_type(4)))  int           v4i;

#define N_TOK  32768
#define DMODEL 2048
#define HD1    1024
#define HD2    512
#define HD3    128
#define NEXP   3

#if defined(__gfx1250__) && __has_builtin(__builtin_amdgcn_tensor_load_to_lds) && \
    __has_builtin(__builtin_amdgcn_s_wait_tensorcnt)
#define USE_TDM 1
#else
#define USE_TDM 0
#endif

// ---------------------------------------------------------------------------
// Transpose + fp32->bf16 convert:  W[K,N] -> Wt[N,K]
// ---------------------------------------------------------------------------
__global__ void k_transpose_cvt(const float* __restrict__ W,
                                __bf16* __restrict__ Wt, int K, int N) {
  int i = blockIdx.x * blockDim.x + threadIdx.x;
  if (i >= K * N) return;
  int k = i / N, n = i % N;
  Wt[(long)n * K + k] = (__bf16)W[i];
}

// ---------------------------------------------------------------------------
// bf16 WMMA GEMM: D[M,N] = relu?(A[M,K] @ Bt[N,K]^T + bias)
// 256 threads = 8 waves; each wave owns a 16x32 C tile (2 WMMAs / A-frag).
// Shared 32n x 32k B panel staged into LDS by the Tensor Data Mover
// (tensor_load_to_lds, wave 0 only, TENSORcnt-tracked), vector-copy fallback.
// ---------------------------------------------------------------------------
template <typename AT>
__global__ __launch_bounds__(256) void k_gemm_bf16(
    const AT* __restrict__ A, const __bf16* __restrict__ Bt,
    const float* __restrict__ bias, __bf16* __restrict__ D,
    int M, int N, int K, int relu) {
  __shared__ __bf16 bt_lds[32 * 32];   // [n 0..31][k 0..31]

  const int lane   = threadIdx.x & 31;
  const int wave   = threadIdx.x >> 5;
  const int tile_n = blockIdx.x * 32;
  const int tile_m = (blockIdx.y * 8 + wave) * 16;

  v8f acc0 = {}, acc1 = {};

  // WMMA 16-bit A layout: lanes 0-15 hold K {0..7,16..23},
  // lanes 16-31 hold K {8..15,24..31}; row = lane & 15.
  const int arow = tile_m + (lane & 15);
  const int kofs = (lane >> 4) << 3;          // 0 or 8

  for (int k0 = 0; k0 < K; k0 += 32) {
    __syncthreads();                           // previous panel reads done
#if USE_TDM
    if (wave == 0) {
      // 2D Tensor DMA descriptor (ISA 8.3/8.4): tile 32(k) x 32(n) bf16
      unsigned lds_addr = (unsigned)(size_t)(&bt_lds[0]);
      unsigned long long ga =
          (unsigned long long)(size_t)(Bt + (size_t)tile_n * K + k0);
      v4u g0;
      g0[0] = 1u;                              // count=1, user mode
      g0[1] = lds_addr;                        // lds_addr
      g0[2] = (unsigned)ga;                    // global_addr[31:0]
      g0[3] = (unsigned)(ga >> 32) | (2u << 30);  // addr[56:32] | type=2
      v8i g1;
      g1[0] = 1 << 16;                         // data_size=2 bytes
      g1[1] = (int)(((unsigned)K & 0xFFFFu) << 16);            // dim0 lo16
      g1[2] = (int)(((unsigned)K >> 16) | (((unsigned)N) << 16));   // dim0 hi | dim1 lo
      g1[3] = (int)(((unsigned)N >> 16) | (32u << 16));        // dim1 hi | tile_dim0=32
      g1[4] = 32;                              // tile_dim1=32, tile_dim2=0
      g1[5] = K;                               // tensor_dim0_stride[31:0]
      g1[6] = 0;                               // stride hi bits
      g1[7] = 0;
      v4i z4 = {};
#if __clang_major__ >= 23
      v8i z8 = {};
      __builtin_amdgcn_tensor_load_to_lds(g0, g1, z4, z4, z8, 0);
#else
      __builtin_amdgcn_tensor_load_to_lds(g0, g1, z4, z4, 0);
#endif
      __builtin_amdgcn_s_wait_tensorcnt(0);
    }
#else
    if (threadIdx.x < 128) {                   // 128 x 16B = 2 KiB panel
      int nr = threadIdx.x >> 2;               // 0..31
      int kk = (threadIdx.x & 3) << 3;         // 0,8,16,24
      *(v8bf*)(bt_lds + nr * 32 + kk) =
          *(const v8bf*)(Bt + (long)(tile_n + nr) * K + k0 + kk);
    }
#endif
    __syncthreads();

    // A fragment (fp32 converted in-register for layer 1)
    v16bf af;
    const AT* ap = A + (long)arow * K + k0 + kofs;
    if (sizeof(AT) == 2) {
      v8bf lo = *(const v8bf*)ap;
      v8bf hi = *(const v8bf*)(ap + 16);
#pragma unroll
      for (int i = 0; i < 8; ++i) { af[i] = lo[i]; af[i + 8] = hi[i]; }
    } else {
#pragma unroll
      for (int i = 0; i < 8; ++i) af[i] = (__bf16)ap[i];
#pragma unroll
      for (int i = 0; i < 8; ++i) af[i + 8] = (__bf16)ap[16 + i];
    }
    __builtin_prefetch(ap + 32, 0, 1);         // speculative: safe past end

    // Two B fragments from the LDS panel (Bt row n = output column)
    v16bf bf0, bf1;
    {
      const __bf16* bp0 = bt_lds + ((lane & 15) << 5) + kofs;
      const __bf16* bp1 = bp0 + (16 << 5);
      v8bf l0 = *(const v8bf*)bp0, h0 = *(const v8bf*)(bp0 + 16);
      v8bf l1 = *(const v8bf*)bp1, h1 = *(const v8bf*)(bp1 + 16);
#pragma unroll
      for (int i = 0; i < 8; ++i) {
        bf0[i] = l0[i]; bf0[i + 8] = h0[i];
        bf1[i] = l1[i]; bf1[i + 8] = h1[i];
      }
    }
    acc0 = __builtin_amdgcn_wmma_f32_16x16x32_bf16(
        false, af, false, bf0, (short)0, acc0, false, false);
    acc1 = __builtin_amdgcn_wmma_f32_16x16x32_bf16(
        false, af, false, bf1, (short)0, acc1, false, false);
  }

  // epilogue: lanes 0-15 -> M=r, lanes 16-31 -> M=8+r; N = lane&15 (+16)
  const int col0  = tile_n + (lane & 15);
  const int rbase = tile_m + ((lane >> 4) << 3);
  const float bv0 = bias[col0];
  const float bv1 = bias[col0 + 16];
#pragma unroll
  for (int r = 0; r < 8; ++r) {
    float v0 = acc0[r] + bv0;
    float v1 = acc1[r] + bv1;
    if (relu) {
      v0 = v0 > 0.0f ? v0 : 0.0f;
      v1 = v1 > 0.0f ? v1 : 0.0f;
    }
    D[(long)(rbase + r) * N + col0]      = (__bf16)v0;
    D[(long)(rbase + r) * N + col0 + 16] = (__bf16)v1;
  }
}

// ---------------------------------------------------------------------------
// Layer 4 (128 -> 3) + softmax + argmax (tiny: VALU is correct here)
// ---------------------------------------------------------------------------
__global__ void k_gate_out(const __bf16* __restrict__ h3,
                           const float* __restrict__ W4,
                           const float* __restrict__ b4,
                           float* __restrict__ probs, int* __restrict__ choice,
                           int n) {
  int row = blockIdx.x * blockDim.x + threadIdx.x;
  if (row >= n) return;
  float a0 = b4[0], a1 = b4[1], a2 = b4[2];
  const __bf16* hp = h3 + (long)row * HD3;
#pragma unroll 4
  for (int k = 0; k < HD3; ++k) {
    float h = (float)hp[k];
    a0 += h * W4[k * 3 + 0];
    a1 += h * W4[k * 3 + 1];
    a2 += h * W4[k * 3 + 2];
  }
  float m  = fmaxf(a0, fmaxf(a1, a2));
  float e0 = __expf(a0 - m), e1 = __expf(a1 - m), e2 = __expf(a2 - m);
  float inv = 1.0f / (e0 + e1 + e2);
  float p0 = e0 * inv, p1 = e1 * inv, p2 = e2 * inv;
  probs[row * 3 + 0] = p0;
  probs[row * 3 + 1] = p1;
  probs[row * 3 + 2] = p2;
  int c = 0; float best = p0;
  if (p1 > best) { best = p1; c = 1; }
  if (p2 > best) { best = p2; c = 2; }
  choice[row] = c;
}

__global__ void k_init(int* counts, int* offsets, float* loss) {
  if (threadIdx.x == 0) {
    counts[0] = counts[1] = counts[2] = counts[3] = 0;
    offsets[0] = offsets[1] = offsets[2] = offsets[3] = 0;
    *loss = 0.0f;
  }
}

__global__ void k_assign(const int* __restrict__ choice, int* counts,
                         int* lists, int n) {
  int i = blockIdx.x * blockDim.x + threadIdx.x;
  if (i >= n) return;
  int e = choice[i];
  int p = atomicAdd(&counts[e], 1);
  lists[e * n + p] = i;
}

// capacity rebalance: every expert gets at least min_per = ceil(n/5);
// steal from the currently-largest overflowing pool's tail.
__global__ void k_rebalance(int* counts, int* lists, int* offsets, int n) {
  if (threadIdx.x != 0 || blockIdx.x != 0) return;
  int min_per = (n + 4) / 5;
  for (int e = 0; e < NEXP; ++e) {
    while (counts[e] < min_per) {
      int d = -1, best = min_per;
      for (int o = 0; o < NEXP; ++o)
        if (o != e && counts[o] > best) { best = counts[o]; d = o; }
      if (d < 0) break;
      int tok = lists[d * n + counts[d] - 1];
      counts[d]--;
      lists[e * n + counts[e]] = tok;
      counts[e]++;
    }
  }
  offsets[0] = 0;
  offsets[1] = counts[0];
  offsets[2] = counts[0] + counts[1];
  offsets[3] = counts[0] + counts[1] + counts[2];
}

// (counts * mean_gate).sum()/3  ==  sum over pooled probs / 3
__global__ void k_loss(const float* __restrict__ probs,
                       const int* __restrict__ lists,
                       const int* __restrict__ counts, float* loss_acc, int n) {
  int e = blockIdx.y;
  int j = blockIdx.x * blockDim.x + threadIdx.x;
  if (j >= counts[e]) return;
  int tok = lists[e * n + j];
  atomicAdd(loss_acc, probs[tok * 3 + e] * (1.0f / 3.0f));
}

// expert_data gather + loss + indices, flattened in return order
__global__ void k_gather(const float* __restrict__ x,
                         const int* __restrict__ lists,
                         const int* __restrict__ offsets,
                         const float* __restrict__ loss_acc,
                         float* __restrict__ out, int n, int d,
                         long out_size) {
  int slot = blockIdx.x;
  int e = (slot >= offsets[2]) ? 2 : (slot >= offsets[1] ? 1 : 0);
  int tok = lists[e * n + (slot - offsets[e])];
  const float4* src = (const float4*)(x + (long)tok * d);
  float4* dst = (float4*)(out + (long)slot * d);
  for (int i = threadIdx.x; i < d / 4; i += blockDim.x) dst[i] = src[i];
  if (threadIdx.x == 0) {
    long base = (long)n * d;
    if (slot == 0 && base < out_size) out[base] = *loss_acc;
    if (base + 1 + slot < out_size) out[base + 1 + slot] = (float)tok;
  }
}

// ---------------------------------------------------------------------------
extern "C" void kernel_launch(void* const* d_in, const int* in_sizes, int n_in,
                              void* d_out, int out_size, void* d_ws,
                              size_t ws_size, hipStream_t stream) {
  const float* x  = (const float*)d_in[0];
  const float* W1 = (const float*)d_in[1];
  const float* b1 = (const float*)d_in[2];
  const float* W2 = (const float*)d_in[3];
  const float* b2 = (const float*)d_in[4];
  const float* W3 = (const float*)d_in[5];
  const float* b3 = (const float*)d_in[6];
  const float* W4 = (const float*)d_in[7];
  const float* b4 = (const float*)d_in[8];
  float* out = (float*)d_out;
  const int n = in_sizes[0] / DMODEL;   // 32768

  // workspace carve-up (256B aligned)
  char* ws = (char*)d_ws;
  size_t o = 0;
  auto take = [&](size_t bytes) {
    void* p = ws + o;
    o += (bytes + 255) & ~(size_t)255;
    return p;
  };
  __bf16* h1    = (__bf16*)take((size_t)n * HD1 * 2);
  __bf16* h2    = (__bf16*)take((size_t)n * HD2 * 2);
  __bf16* h3    = (__bf16*)take((size_t)n * HD3 * 2);
  __bf16* w1t   = (__bf16*)take((size_t)DMODEL * HD1 * 2);
  __bf16* w2t   = (__bf16*)take((size_t)HD1 * HD2 * 2);
  __bf16* w3t   = (__bf16*)take((size_t)HD2 * HD3 * 2);
  float*  probs = (float*)take((size_t)n * 3 * 4);
  int*    choice  = (int*)take((size_t)n * 4);
  int*    lists   = (int*)take((size_t)3 * n * 4);
  int*    counts  = (int*)take(16 * 4);
  int*    offsets = (int*)take(16 * 4);
  float*  loss    = (float*)take(16 * 4);

  k_init<<<1, 64, 0, stream>>>(counts, offsets, loss);

  k_transpose_cvt<<<(DMODEL * HD1 + 255) / 256, 256, 0, stream>>>(W1, w1t, DMODEL, HD1);
  k_transpose_cvt<<<(HD1 * HD2 + 255) / 256, 256, 0, stream>>>(W2, w2t, HD1, HD2);
  k_transpose_cvt<<<(HD2 * HD3 + 255) / 256, 256, 0, stream>>>(W3, w3t, HD2, HD3);

  // gate MLP on the matrix pipe (each wave: 16x32 C tile; block: 32 N cols)
  k_gemm_bf16<float><<<dim3(HD1 / 32, n / 128), 256, 0, stream>>>(
      x, w1t, b1, h1, n, HD1, DMODEL, 1);
  k_gemm_bf16<__bf16><<<dim3(HD2 / 32, n / 128), 256, 0, stream>>>(
      h1, w2t, b2, h2, n, HD2, HD1, 1);
  k_gemm_bf16<__bf16><<<dim3(HD3 / 32, n / 128), 256, 0, stream>>>(
      h2, w3t, b3, h3, n, HD3, HD2, 1);

  k_gate_out<<<(n + 255) / 256, 256, 0, stream>>>(h3, W4, b4, probs, choice, n);

  k_assign<<<(n + 255) / 256, 256, 0, stream>>>(choice, counts, lists, n);
  k_rebalance<<<1, 1, 0, stream>>>(counts, lists, offsets, n);
  k_loss<<<dim3((n + 255) / 256, 3), 256, 0, stream>>>(probs, lists, counts, loss, n);

  k_gather<<<n, 256, 0, stream>>>(x, lists, offsets, loss, out, n, DMODEL,
                                  (long)out_size);
}
